// WrnYolov2_73203422593277
// MI455X (gfx1250) — matrix operand order repaired and checked by reading, not compiled, not run on hardware
//
#include <hip/hip_runtime.h>

typedef __bf16 bf16_t;
typedef __attribute__((ext_vector_type(16))) __bf16 v16bf;
typedef __attribute__((ext_vector_type(8)))  __bf16 v8bf;
typedef __attribute__((ext_vector_type(8)))  float  v8f;
typedef __attribute__((ext_vector_type(4)))  unsigned u32x4;

// Geometry
// stage5: [32,512,38,38]  stage6: [32,1024,19,19]
// xin (NHWC bf16): [32*361][1280]  (ch 0..255 = reorg(conv1), 256..1279 = stage6)
// conv2 GEMM: M=1024, N=11552, K=11520 (k = tap*1280 + ci, tap = ky*3+kx)
// head  GEMM: M=600 (pad 608), N=11552, K=1024

#define KS_C2  360            // 11520/32
#define KS_HD  32             // 1024/32
#define NPIXT  11552          // 32*361

// ---------------- Stage 1: 1x1 conv + BN + leaky + reorg -> xin[ch 0..255]
// block: 64 c x 4 w; stage5 channel chunks staged through LDS (read once per block)
__global__ void __launch_bounds__(256) k_conv1_reorg(const float* __restrict__ s5, const float* __restrict__ w1,
                              const float* __restrict__ g1, const float* __restrict__ b1,
                              const float* __restrict__ m1, const float* __restrict__ v1,
                              bf16_t* __restrict__ xin) {
    __shared__ float ls[256];                        // [64 ci][4 w]
    int bid = blockIdx.x;                            // 32*38*10 blocks
    int wb = bid % 10; int tmp = bid / 10;
    int h = tmp % 38; int b = tmp / 38;
    int tid = threadIdx.x;
    int c = tid >> 2, wq = tid & 3;
    int w0 = wb * 4;
    int wld = w0 + wq; if (wld > 37) wld = 37;       // clamped for loads

    const float* s5base = s5 + ((size_t)b * 512) * 1444 + h * 38;
    const float* wrow = w1 + c * 512;

    float acc = 0.f;
#pragma unroll
    for (int cib = 0; cib < 8; ++cib) {
        __syncthreads();
        ls[tid] = s5base[(size_t)(cib * 64 + (tid >> 2)) * 1444 + wld];
        __syncthreads();
        const float* wc = wrow + cib * 64;
#pragma unroll 16
        for (int cc = 0; cc < 64; ++cc)
            acc += ls[cc * 4 + wq] * wc[cc];
    }

    if (w0 + wq < 38) {
        float scale = g1[c] * rsqrtf(v1[c] + 1e-5f);
        float shift = b1[c] - m1[c] * scale;
        float y = acc * scale + shift;
        y = (y > 0.f) ? y : 0.1f * y;
        int w = w0 + wq;
        int ho = h >> 1, sy = h & 1, wo = w >> 1, sx = w & 1;
        int rc = (sy * 2 + sx) * 64 + c;
        xin[((size_t)b * 361 + ho * 19 + wo) * 1280 + rc] = (bf16_t)y;
    }
}

// ---------------- Stage 2: stage6 -> bf16, channels 256..1279 (concat by layout)
__global__ void k_cast6(const float* __restrict__ s6, bf16_t* __restrict__ xin) {
    int t = blockIdx.x * 256 + threadIdx.x;          // (b*361+pix)*1024 + c, exact
    int c = t & 1023; int np = t >> 10;
    int b = np / 361; int pix = np - b * 361;
    float v = s6[((size_t)(b * 1024 + c)) * 361 + pix];
    xin[(size_t)np * 1280 + 256 + c] = (bf16_t)v;
}

// A-fragment K index for (half h, lane-group g): ISA 16-bit A 16x32 layout
__device__ __forceinline__ int a_kidx(int g, int h) {
    return h + 8 * g + (h >= 8 ? 8 : 0);
}

// ---------------- Weight repack: w2*bn_scale -> WMMA A fragments [mt64][ks360][lane32][h16]
__global__ void k_packw2(const float* __restrict__ w2, const float* __restrict__ g2,
                         const float* __restrict__ v2, bf16_t* __restrict__ wpack) {
    int t = blockIdx.x * 256 + threadIdx.x;          // exact 64*360*512
    int h = t & 15; int lane = (t >> 4) & 31;
    int ks = (t >> 9) % KS_C2; int mt = t / (512 * KS_C2);
    int m = mt * 16 + (lane & 15); int g = lane >> 4;
    int k = ks * 32 + a_kidx(g, h);
    int tap = k / 1280; int ci = k - tap * 1280;
    int ky = tap / 3, kx = tap - ky * 3;
    float scale = g2[m] * rsqrtf(v2[m] + 1e-5f);
    float val = w2[(((size_t)m * 1280 + ci) * 3 + ky) * 3 + kx] * scale;
    wpack[t] = (bf16_t)val;
}

// ---------------- meta_state cls weights -> A fragments [mt38][ks32][lane32][h16]
__global__ void k_packmeta(const float* __restrict__ meta, bf16_t* __restrict__ apack) {
    int t = blockIdx.x * 256 + threadIdx.x;          // exact 38*32*512
    int h = t & 15; int lane = (t >> 4) & 31;
    int ks = (t >> 9) & 31; int mt = t >> 14;
    int r = mt * 16 + (lane & 15); int g = lane >> 4;
    int k = ks * 32 + a_kidx(g, h);
    float val = (r < 600) ? meta[(size_t)r * 1025 + k] : 0.f;
    apack[t] = (bf16_t)val;
}

// ---------------- Stage 3: 3x3 conv as implicit GEMM with bf16 WMMA
// block = 256 thr = 8 waves; each wave: 4 M-tiles x 4 N-tiles -> 16 WMMA per k-step
// block: 512 out ch x 64 pixels; distance-2 fill pipeline, double-buffered LDS
__global__ void __launch_bounds__(256) k_conv2(const bf16_t* __restrict__ xin,
                                               const bf16_t* __restrict__ wpack,
                                               const float* __restrict__ g2, const float* __restrict__ b2,
                                               const float* __restrict__ m2, const float* __restrict__ v2,
                                               bf16_t* __restrict__ pre) {
    __shared__ alignas(64) bf16_t lb[2][4][32 * 16]; // [buf][ntile][lane][16 k elems], 8KB
    int ngrp = blockIdx.x;                           // 0..180 (64 pixels each; tail clamped)
    int mblk = blockIdx.y;                           // 0..1
    int tid = threadIdx.x, lane = tid & 31, wv = tid >> 5;
    int mt0 = mblk * 32 + wv * 4;                    // 4 M-tiles per wave

    // fill role: pixel p = tid>>2 (0..63), chunk q = tid&3 (k' = 8q..8q+7, one b128)
    int p = tid >> 2, q = tid & 3;
    int jt = p >> 4, pl = p & 15;                    // LDS tile index, column
    int ng = ngrp * 64 + p;
    if (ng > NPIXT - 1) ng = NPIXT - 1;              // clamp tail (stores guarded)

    int b = ng / 361, pp = ng - b * 361, y = pp / 19, x = pp - y * 19;

    unsigned vmask = 0;
#pragma unroll
    for (int tap = 0; tap < 9; ++tap) {
        int yy = y + tap / 3 - 1, xx = x + tap % 3 - 1;
        if (yy >= 0 && yy < 19 && xx >= 0 && xx < 19) vmask |= 1u << tap;
    }
    const bf16_t* base = xin + ((size_t)b * 361 + y * 19 + x) * 1280 + 8 * q;

    auto fetch = [&](int ks, u32x4& d, bool& v) {
        int tap = (ks * 205) >> 13;                  // ks/40 for ks<360
        int t3  = (tap * 11) >> 5;                   // tap/3 for tap<9
        int off = (t3 * 19 + (tap - t3 * 3) - 20) * 1280 + (ks - tap * 40) * 32;
        v = (vmask >> tap) & 1;
        d = *(const u32x4*)(v ? base + off : base);  // safe fallback addr
    };
    auto stor = [&](int bufi, u32x4 d, bool v) {
        union { u32x4 u; bf16_t h[8]; } cv;
        cv.u = v ? d : (u32x4){0, 0, 0, 0};          // select at store time
        bf16_t* dst = &lb[bufi][jt][(pl + 16 * (q & 1)) * 16 + 8 * (q >> 1)];
#pragma unroll
        for (int i = 0; i < 8; ++i) dst[i] = cv.h[i];
    };

    v8f acc[4][4];
#pragma unroll
    for (int i = 0; i < 4; ++i)
#pragma unroll
        for (int j = 0; j < 4; ++j) acc[i][j] = (v8f){};

    const v16bf* wp = (const v16bf*)wpack;

    // pipeline prologue: data(0) stored, data(1) in flight
    u32x4 d1; bool v1f;
    {
        u32x4 d0; bool v0f;
        fetch(0, d0, v0f);
        stor(0, d0, v0f);
        fetch(1, d1, v1f);
    }
    __syncthreads();

#pragma unroll 2
    for (int ks = 0; ks < KS_C2; ++ks) {
        int cur = ks & 1;
        u32x4 d2 = (u32x4){0, 0, 0, 0}; bool v2f = false;
        if (ks + 2 < KS_C2) fetch(ks + 2, d2, v2f);  // distance-2 fetch

        v16bf bf[4];
#pragma unroll
        for (int j = 0; j < 4; ++j)
            bf[j] = *(const v16bf*)&lb[cur][j][lane * 16];

        if (ks + 4 < KS_C2) {                        // gfx1250 prefetch path
            __builtin_prefetch(&wp[((size_t)(mt0 + 0) * KS_C2 + ks + 4) * 32 + lane], 0, 1);
            __builtin_prefetch(&wp[((size_t)(mt0 + 2) * KS_C2 + ks + 4) * 32 + lane], 0, 1);
        }
#pragma unroll
        for (int i = 0; i < 4; ++i) {
            v16bf ai = wp[((size_t)(mt0 + i) * KS_C2 + ks) * 32 + lane];
#pragma unroll
            for (int j = 0; j < 4; ++j)
                acc[i][j] = __builtin_amdgcn_wmma_f32_16x16x32_bf16(
                    false, ai, false, bf[j], (short)0, acc[i][j], false, false);
        }

        if (ks + 1 < KS_C2) stor(cur ^ 1, d1, v1f);
        d1 = d2; v1f = v2f;
        __syncthreads();                             // one barrier per k-step
    }

    // epilogue: + BN shift, leaky, vectorized bf16 store (16B per 8 channels)
    int nl = lane & 15, hi = lane >> 4;
#pragma unroll
    for (int i = 0; i < 4; ++i) {
        int mt = mt0 + i;
        float shift8[8];
#pragma unroll
        for (int v = 0; v < 8; ++v) {
            int co = mt * 16 + v + 8 * hi;
            float sc = g2[co] * rsqrtf(v2[co] + 1e-5f);
            shift8[v] = b2[co] - m2[co] * sc;        // scale folded into weights
        }
#pragma unroll
        for (int j = 0; j < 4; ++j) {
            int ngj = ngrp * 64 + j * 16 + nl;
            if (ngj < NPIXT) {
                v8bf ov;
#pragma unroll
                for (int v = 0; v < 8; ++v) {
                    float f = acc[i][j][v] + shift8[v];
                    f = (f > 0.f) ? f : 0.1f * f;
                    ov[v] = (bf16_t)f;
                }
                *(v8bf*)&pre[(size_t)ngj * 1024 + mt * 16 + 8 * hi] = ov;
            }
        }
    }
}

// ---------------- Stage 4: head GEMM (600x11552x1024) + bias, f32 out
__global__ void __launch_bounds__(256) k_head(const bf16_t* __restrict__ pre,
                                              const bf16_t* __restrict__ apack,
                                              const float* __restrict__ meta,
                                              float* __restrict__ out) {
    __shared__ alignas(64) bf16_t lb[2][32 * 16];
    int ntile = blockIdx.x, mblk = blockIdx.y;
    int tid = threadIdx.x, lane = tid & 31, wv = tid >> 5;
    int mt = mblk * 8 + wv;                          // 0..39, valid < 38

    int p = tid >> 4, q = tid & 15;
    int ngp = ntile * 16 + p;
    const bf16_t* src = pre + (size_t)ngp * 1024 + 2 * q;

    auto stor = [&](int bufi, unsigned d) {
        union { unsigned u; bf16_t h[2]; } cv; cv.u = d;
        lb[bufi][(p + 16 * ((q >> 2) & 1)) * 16 + 2 * (q & 3) + 8 * (q >> 3)]     = cv.h[0];
        lb[bufi][(p + 16 * ((q >> 2) & 1)) * 16 + 2 * (q & 3) + 8 * (q >> 3) + 1] = cv.h[1];
    };

    v8f acc = {};
    const v16bf* pA = (const v16bf*)apack + ((size_t)mt * KS_HD) * 32 + lane;

    unsigned d1;
    stor(0, *(const unsigned*)src);
    d1 = *(const unsigned*)(src + 32);
    __syncthreads();

#pragma unroll 2
    for (int ks = 0; ks < KS_HD; ++ks) {
        int cur = ks & 1;
        unsigned d2 = 0;
        if (ks + 2 < KS_HD) d2 = *(const unsigned*)(src + (ks + 2) * 32);
        if (mt < 38) {
            v16bf bfrag = *(const v16bf*)&lb[cur][lane * 16];
            v16bf a = pA[(size_t)ks * 32];
            acc = __builtin_amdgcn_wmma_f32_16x16x32_bf16(false, a, false, bfrag, (short)0, acc, false, false);
        }
        if (ks + 1 < KS_HD) stor(cur ^ 1, d1);
        d1 = d2;
        __syncthreads();
    }

    if (mt < 38) {
        int nl = lane & 15, hi = lane >> 4;
        int ng = ntile * 16 + nl;
        int b = ng / 361, pix = ng - b * 361;
#pragma unroll
        for (int v = 0; v < 8; ++v) {
            int r = mt * 16 + v + 8 * hi;
            if (r < 600) {
                float bias = meta[(size_t)r * 1025 + 1024];
                int c = r / 30, m = r - c * 30;
                out[(((size_t)(b * 20 + c)) * 30 + m) * 361 + pix] = acc[v] + bias;
            }
        }
    }
}

extern "C" void kernel_launch(void* const* d_in, const int* in_sizes, int n_in,
                              void* d_out, int out_size, void* d_ws, size_t ws_size,
                              hipStream_t stream) {
    (void)in_sizes; (void)n_in; (void)out_size; (void)ws_size;
    const float* s6  = (const float*)d_in[0];
    const float* s5  = (const float*)d_in[1];
    const float* w1  = (const float*)d_in[2];
    const float* g1  = (const float*)d_in[3];
    const float* b1  = (const float*)d_in[4];
    const float* m1  = (const float*)d_in[5];
    const float* v1  = (const float*)d_in[6];
    const float* w2  = (const float*)d_in[7];
    const float* g2  = (const float*)d_in[8];
    const float* b2  = (const float*)d_in[9];
    const float* m2  = (const float*)d_in[10];
    const float* v2  = (const float*)d_in[11];
    const float* meta= (const float*)d_in[12];

    char* ws = (char*)d_ws;
    bf16_t* xin   = (bf16_t*)(ws);                     // 11552*1280*2 = 29,573,120 B
    bf16_t* wpack = (bf16_t*)(ws + 29573120);          // 64*360*512*2 = 23,592,960 B
    bf16_t* pre   = (bf16_t*)(ws + 53166080);          // 11552*1024*2 = 23,658,496 B
    bf16_t* apack = (bf16_t*)(ws + 76824576);          // 38*32*512*2  =  1,245,184 B
    float* out = (float*)d_out;

    k_conv1_reorg<<<12160, 256, 0, stream>>>(s5, w1, g1, b1, m1, v1, xin);
    k_cast6     <<<46208, 256, 0, stream>>>(s6, xin);
    k_packw2    <<<46080, 256, 0, stream>>>(w2, g2, v2, wpack);
    k_packmeta  <<<2432,  256, 0, stream>>>(meta, apack);
    k_conv2     <<<dim3(181, 2), 256, 0, stream>>>(xin, wpack, g2, b2, m2, v2, pre);
    k_head      <<<dim3(722, 5), 256, 0, stream>>>(pre, apack, meta, out);
}
